// BiLSTMw2v_64665027609092
// MI455X (gfx1250) — matrix operand order repaired and checked by reading, not compiled, not run on hardware
//
#include <hip/hip_runtime.h>
#include <hip/hip_bf16.h>

// ---------------------------------------------------------------------------
// BiLSTM word-tagger for MI455X (gfx1250, wave32, WMMA).
//
//   1) embed_relu : sent[t, 0:320] = bf16(relu(emb[x[t]])), zero-padded K
//   2) prep_whh   : Whh_{f,b} f32 -> bf16 (fits in 320KB LDS)
//   3) xproj_gemm : x_proj[dir] = sent @ Wih[dir].T + bih + bhh   (WMMA bf16)
//   4) lstm_scan  : 2 persistent workgroups (one per direction); Whh LDS-
//                   resident; per step the 1x200 x 200x800 recurrent GEMV is
//                   done as 50x7 v_wmma_f32_16x16x32_bf16 tiles (A row 0 = h).
//   5) head       : out = relu(hcat @ W_h2s + b) @ W_s2o + b
// ---------------------------------------------------------------------------

#define TT   4096
#define EE   300
#define EK   320      // input-proj K padded to multiple of 32
#define HH   200
#define KP   224      // recurrent K padded to 7 tiles of 32
#define G4   800      // 4*H
#define XHH  50
#define OO   2

typedef __attribute__((ext_vector_type(16))) __bf16 bf16x16;
typedef __attribute__((ext_vector_type(8)))  float  floatx8;

// --------------------------- Phase 1: embedding -----------------------------
__global__ void embed_relu(const int* __restrict__ x,
                           const float* __restrict__ emb,
                           __bf16* __restrict__ sent) {
  const int t = blockIdx.x;          // 0..T-1
  const int c = threadIdx.x;         // 0..319
  const int idx = x[t];
  float v = 0.0f;
  if (c < EE) v = emb[(size_t)idx * EE + c];
  v = fmaxf(v, 0.0f);
  sent[(size_t)t * EK + c] = (__bf16)v;
}

// ------------------- Phase 2: Whh f32 -> bf16 conversion --------------------
__global__ void prep_whh(const float* __restrict__ whhf,
                         const float* __restrict__ whhb,
                         __bf16* __restrict__ wf,
                         __bf16* __restrict__ wb) {
  const int i = blockIdx.x * 256 + threadIdx.x;
  if (i < G4 * HH) {
    wf[i] = (__bf16)whhf[i];
    wb[i] = (__bf16)whhb[i];
  }
}

// ----------------- Phase 3: input projection GEMM (WMMA) --------------------
// One wave per 16x16 tile of x_proj[dir] (M=4096, N=800, K=320 padded).
__global__ __launch_bounds__(32)
void xproj_gemm(const __bf16* __restrict__ sent,
                const float* __restrict__ WihF, const float* __restrict__ bihF,
                const float* __restrict__ bhhF,
                const float* __restrict__ WihB, const float* __restrict__ bihB,
                const float* __restrict__ bhhB,
                float* __restrict__ xproj) {   // [2, T, 800]
  const int lane  = threadIdx.x;               // 0..31
  const int mtile = blockIdx.x;                // 0..255
  const int ntile = blockIdx.y;                // 0..49
  const int dir   = blockIdx.z;                // 0..1
  const bool hi   = (lane >> 4) != 0;
  const int  l15  = lane & 15;

  const float* __restrict__ W  = dir ? WihB : WihF;
  const float* __restrict__ bi = dir ? bihB : bihF;
  const float* __restrict__ bh = dir ? bhhB : bhhF;

  const int row = mtile * 16 + l15;            // A row (M)
  const int nn  = ntile * 16 + l15;            // B col (N)

  floatx8 acc = {};
  for (int k0 = 0; k0 < EK; k0 += 32) {
    // A tile 16x32 bf16: lane<16: K=k0+0..15
    //                    lane>=16: K=k0+16..31
    bf16x16 a;
    const __bf16* ap = sent + (size_t)row * EK + k0 + (hi ? 16 : 0);
    ((uint4*)&a)[0] = *(const uint4*)(ap);        // halves 0..7
    ((uint4*)&a)[1] = *(const uint4*)(ap + 8);    // halves 8..15

    // B tile 32x16: half-slot j of lane -> B[kb+j][nn]; B[k][n] = Wih[n*E + k]
    bf16x16 b;
    const int kb = k0 + (hi ? 16 : 0);
    const float* wrow = W + (size_t)nn * EE;
#pragma unroll
    for (int j = 0; j < 16; ++j) {
      const int k = kb + j;
      const int kc = (k < EE) ? k : (EE - 1);    // clamp address, mask value
      float wv = wrow[kc];
      b[j] = (k < EE) ? (__bf16)wv : (__bf16)0.0f;
    }

    acc = __builtin_amdgcn_wmma_f32_16x16x32_bf16(
        false, a, false, b, (short)0, acc, false, false);
  }

  // D 16x16 f32: VGPR r -> (M = r or r+8, N = lane&15). Fold bias.
  const float bias  = bi[nn] + bh[nn];
  const int   mbase = mtile * 16 + (hi ? 8 : 0);
  float* __restrict__ out = xproj + (size_t)dir * TT * G4;
#pragma unroll
  for (int r = 0; r < 8; ++r)
    out[(size_t)(mbase + r) * G4 + nn] = acc[r] + bias;
}

// -------------------- Phase 4: sequential LSTM scan (WMMA) ------------------
// One workgroup per direction, 256 threads = 8 waves. Whh bf16 [800][200]
// lives in LDS. Per step: gates(1x800) = h(1x200) x Whh^T as 50 N-tiles x
// 7 K-tiles of v_wmma_f32_16x16x32_bf16; A has h in row 0, rows 1..15 zero.
// B tile = 2x ds_load_b128 per lane (16 consecutive k of one Whh row).
__device__ __forceinline__ float sigmoid_f(float v) {
  return 1.0f / (1.0f + __expf(-v));
}

__global__ __launch_bounds__(256)
void lstm_scan(const __bf16* __restrict__ whhF,
               const __bf16* __restrict__ whhB,
               const float* __restrict__ xproj,   // [2, T, 800]
               float* __restrict__ hcat) {        // [T, 400]
  extern __shared__ __align__(16) char smem[];
  __bf16* wl    = (__bf16*)smem;                  // [800*200]      320,000 B
  __bf16* hb    = (__bf16*)(smem + 320000);       // [224] h (bf16)     512 B
  float*  gates = (float*) (smem + 320512);       // [800]            3,200 B
                                                  // total          323,712 B
  const int dir  = blockIdx.x;
  const int tid  = threadIdx.x;                   // 0..255
  const int lane = tid & 31;
  const int wave = tid >> 5;                      // 0..7
  const bool hi  = (lane >> 4) != 0;
  const int  l15 = lane & 15;

  // ---- one-time: stage Whh (bf16) into LDS as uint4 copies
  const uint4* __restrict__ wsrc4 = (const uint4*)(dir ? whhB : whhF);
  uint4* wl4 = (uint4*)wl;
  for (int i = tid; i < (G4 * HH) / 8; i += 256) wl4[i] = wsrc4[i];
  for (int i = tid; i < KP; i += 256) hb[i] = (__bf16)0.0f;  // pads k>=200 too
  __syncthreads();

  float c_reg = 0.0f;                             // cell state (tid < 200)

  const float* __restrict__ xp   = xproj + (size_t)dir * TT * G4;
  float* __restrict__       hout = hcat + dir * HH;   // row stride 2H

  for (int t = 0; t < TT; ++t) {
    if (t + 1 < TT) __builtin_prefetch(xp + (size_t)(t + 1) * G4 + tid, 0, 0);

    // ---- A fragments for all 7 K-tiles: row M=0 holds h, rows 1..15 zero.
    //      Only lanes 0 (K=kt*32+0..15) and 16 (K=kt*32+16..31)
    //      carry data; hb[200..223] is zero so no masking needed.
    bf16x16 afrag[7];
#pragma unroll
    for (int kt = 0; kt < 7; ++kt) {
      bf16x16 a = {};
      if (l15 == 0) {
        const int kb = kt * 32 + (hi ? 16 : 0);
        ((uint4*)&a)[0] = *(const uint4*)(hb + kb);        // halves 0..7
        ((uint4*)&a)[1] = *(const uint4*)(hb + kb + 8);    // halves 8..15
      }
      afrag[kt] = a;
    }

    // ---- per-wave N-tiles (wave-uniform loop; EXEC all-1s at each WMMA)
    for (int nt = wave; nt < G4 / 16; nt += 8) {
      floatx8 acc = {};
      const int n = nt * 16 + l15;
      const __bf16* wrow = wl + (size_t)n * HH;            // Whh row n
#pragma unroll
      for (int kt = 0; kt < 7; ++kt) {
        bf16x16 b;
        if (kt < 6) {
          const int kb = kt * 32 + (hi ? 16 : 0);
          ((uint4*)&b)[0] = *(const uint4*)(wrow + kb);      // k=kb..kb+7
          ((uint4*)&b)[1] = *(const uint4*)(wrow + kb + 8);  // k=kb+8..kb+15
        } else {
          bf16x16 bz = {};
          if (!hi)                                           // k=192..199 only
            ((uint4*)&bz)[0] = *(const uint4*)(wrow + 192);
          b = bz;                                            // k>=200 -> 0
        }
        acc = __builtin_amdgcn_wmma_f32_16x16x32_bf16(
            false, afrag[kt], false, b, (short)0, acc, false, false);
      }
      // D row 0 = VGPR0, lanes 0..15 (M=0, N=lane)
      if (!hi) gates[nt * 16 + l15] = acc[0] + xp[(size_t)t * G4 + nt * 16 + l15];
    }
    __syncthreads();

    // ---- LSTM elementwise update; c in registers, h back to LDS as bf16
    if (tid < HH) {
      const float ig = sigmoid_f(gates[tid]);
      const float fg = sigmoid_f(gates[HH + tid]);
      const float gg = tanhf(gates[2 * HH + tid]);
      const float og = sigmoid_f(gates[3 * HH + tid]);
      c_reg = fg * c_reg + ig * gg;
      const float hn = og * tanhf(c_reg);
      hb[tid] = (__bf16)hn;
      hout[(size_t)t * (2 * HH) + tid] = hn;     // fire-and-forget global store
    }
    __syncthreads();
  }
}

// ------------------------- Phase 5: output head -----------------------------
__global__ __launch_bounds__(64)
void head(const float* __restrict__ hcat,      // [T, 400]
          const float* __restrict__ Wh2s,      // [400, 50]
          const float* __restrict__ bh2s,      // [50]
          const float* __restrict__ Ws2o,      // [50, 2]
          const float* __restrict__ bs2o,      // [2]
          float* __restrict__ out) {           // [T, 2]
  __shared__ float row[2 * HH];
  __shared__ float s[XHH];
  const int t = blockIdx.x;
  const int tid = threadIdx.x;                 // 0..63
  for (int i = tid; i < 2 * HH; i += 64) row[i] = hcat[(size_t)t * (2 * HH) + i];
  __syncthreads();
  if (tid < XHH) {
    float acc = bh2s[tid];
#pragma unroll 4
    for (int k = 0; k < 2 * HH; ++k) acc += row[k] * Wh2s[(size_t)k * XHH + tid];
    s[tid] = fmaxf(acc, 0.0f);
  }
  __syncthreads();
  if (tid < OO) {
    float acc = bs2o[tid];
#pragma unroll
    for (int x = 0; x < XHH; ++x) acc += s[x] * Ws2o[(size_t)x * OO + tid];
    out[(size_t)t * OO + tid] = acc;
  }
}

// ---------------------------------------------------------------------------
extern "C" void kernel_launch(void* const* d_in, const int* in_sizes, int n_in,
                              void* d_out, int out_size, void* d_ws, size_t ws_size,
                              hipStream_t stream) {
  (void)in_sizes; (void)n_in; (void)out_size; (void)ws_size;

  const int*   x     = (const int*)  d_in[0];
  const float* emb   = (const float*)d_in[1];
  const float* WihF  = (const float*)d_in[2];
  const float* WhhF  = (const float*)d_in[3];
  const float* bihF  = (const float*)d_in[4];
  const float* bhhF  = (const float*)d_in[5];
  const float* WihB  = (const float*)d_in[6];
  const float* WhhB  = (const float*)d_in[7];
  const float* bihB  = (const float*)d_in[8];
  const float* bhhB  = (const float*)d_in[9];
  const float* Wh2s  = (const float*)d_in[10];
  const float* bh2s  = (const float*)d_in[11];
  const float* Ws2o  = (const float*)d_in[12];
  const float* bs2o  = (const float*)d_in[13];
  float* out = (float*)d_out;

  // ---- workspace layout (bytes, 256-aligned); total ~36.03 MB ----
  char* ws = (char*)d_ws;
  __bf16* sent   = (__bf16*)(ws + 0);                       //  2,621,440
  float*  xproj  = (float*) (ws + 2621440);                 // 26,214,400
  __bf16* whhFb  = (__bf16*)(ws + 28835840);                //    320,000
  __bf16* whhBb  = (__bf16*)(ws + 29155840);                //    320,000
  float*  hcat   = (float*) (ws + 29475840);                //  6,553,600

  embed_relu<<<TT, EK, 0, stream>>>(x, emb, sent);

  prep_whh<<<(G4 * HH + 255) / 256, 256, 0, stream>>>(WhhF, WhhB, whhFb, whhBb);

  xproj_gemm<<<dim3(TT / 16, G4 / 16, 2), 32, 0, stream>>>(
      sent, WihF, bihF, bhhF, WihB, bihB, bhhB, xproj);

  // 2 workgroups (fwd/bwd), 323,712 B dynamic LDS each (<320KB WGP limit)
  lstm_scan<<<2, 256, 323712, stream>>>(whhFb, whhBb, xproj, hcat);

  head<<<TT, 64, 0, stream>>>(hcat, Wh2s, bh2s, Ws2o, bs2o, out);
}
